// GemmaAttention_7980049236174
// MI455X (gfx1250) — matrix operand order repaired and checked by reading, not compiled, not run on hardware
//
#include <hip/hip_runtime.h>
#include <hip/hip_bf16.h>
#include <cstdint>
#include <cstddef>

// ---- problem constants (match reference) ----
#define H_    8
#define KV_   1
#define D_    256
#define HID_  2048
#define B_    4
#define S_    2048
#define T_    (B_ * S_)              // 8192
#define NQKV_ ((H_ + 2 * KV_) * D_) // 2560
#define NSLOTS_ 16384

typedef __attribute__((ext_vector_type(16))) __bf16 v16bf;
typedef __attribute__((ext_vector_type(8)))  __bf16 v8bf;
typedef __attribute__((ext_vector_type(8)))  float  v8f;
typedef __attribute__((ext_vector_type(4)))  int    v4i;

#if defined(__gfx1250__) && __has_builtin(__builtin_amdgcn_global_load_async_to_lds_b128)
#define HAVE_ASYNC_LDS 1
#else
#define HAVE_ASYNC_LDS 0
#endif

// 16-byte global -> LDS copy: async (ASYNCcnt-tracked) when available.
// Builtin signature (probe-confirmed by round-2 diagnostic):
//   (v4i addrspace(1)*, v4i addrspace(3)*, imm offset, imm cpol)
__device__ inline void copy16_g2l(const __bf16* g, __bf16* l) {
#if HAVE_ASYNC_LDS
    __builtin_amdgcn_global_load_async_to_lds_b128(
        (__attribute__((address_space(1))) v4i*)g,
        (__attribute__((address_space(3))) v4i*)l, 0, 0);
#else
    *(v8bf*)l = *(const v8bf*)g;
#endif
}

// ---------- fragment helpers ----------
// A-operand (16x32 bf16): lane L: m=L%16, elems 0..15 -> K = 16*half + e,
// half = L/16.
__device__ inline v16bf a_frag(const __bf16* row_base, int half) {
    v8bf lo = *(const v8bf*)(row_base + 16 * half);
    v8bf hi = *(const v8bf*)(row_base + 16 * half + 8);
    v16bf r;
#pragma unroll
    for (int e = 0; e < 8; ++e) { r[e] = lo[e]; r[8 + e] = hi[e]; }
    return r;
}

// B-operand (32xN bf16): lane L: n=L%16, elems 0..15 -> K = 16*half + e.
__device__ inline v16bf b_frag(const __bf16* col_base, int half) {
    v8bf lo = *(const v8bf*)(col_base + 16 * half);
    v8bf hi = *(const v8bf*)(col_base + 16 * half + 8);
    v16bf r;
#pragma unroll
    for (int e = 0; e < 8; ++e) { r[e] = lo[e]; r[8 + e] = hi[e]; }
    return r;
}

// ---------- fp32 -> bf16 pre-conversion ----------
__global__ __launch_bounds__(256) void cvt_bf16_kernel(
    const float* __restrict__ src, __bf16* __restrict__ dst, int n) {
    int i = (blockIdx.x * blockDim.x + threadIdx.x) * 4;
    if (i >= n) return;
    float4 v = *(const float4*)(src + i);
    dst[i + 0] = (__bf16)v.x;
    dst[i + 1] = (__bf16)v.y;
    dst[i + 2] = (__bf16)v.z;
    dst[i + 3] = (__bf16)v.w;
}

// ---------- GEMM: C[M,N](f32) = A[M,K] * B[N,K]^T, bf16 WMMA ----------
// 256 thr = 8 waves (4 M x 2 N); wave tile 32x64; block tile 128x128.
// A/B K-slabs (128x32) double-buffered in LDS via async global->LDS copies.
#define LDS_STRIDE 40  // 32 + 8 pad (80B rows -> conflict-free b128 reads)
__global__ __launch_bounds__(256) void gemm_nt_bf16_kernel(
    const __bf16* __restrict__ Ab, const __bf16* __restrict__ Bb,
    float* __restrict__ Cp, int M, int N, int K) {
    __shared__ alignas(16) __bf16 As[2][128][LDS_STRIDE];
    __shared__ alignas(16) __bf16 Bs[2][128][LDS_STRIDE];

    const int tid  = threadIdx.x;
    const int lane = tid & 31;
    const int w    = tid >> 5;
    const int half = lane >> 4;
    const int l16  = lane & 15;
    const int wm = w >> 1;  // 0..3
    const int wn = w & 1;   // 0..1
    const int m0 = blockIdx.y * 128;
    const int n0 = blockIdx.x * 128;

    // Each thread copies 2 x 16B for A and 2 x 16B for B per K-slab.
    // chunk c in [0,512): row = c>>2, 16B sub-chunk = c&3.
    auto issue_slab = [&](int buf, int k0) {
#pragma unroll
        for (int q = 0; q < 2; ++q) {
            const int c = 2 * tid + q;
            const int row = c >> 2;
            const int sub = c & 3;
            copy16_g2l(Ab + (size_t)(m0 + row) * K + k0 + sub * 8,
                       &As[buf][row][sub * 8]);
            copy16_g2l(Bb + (size_t)(n0 + row) * K + k0 + sub * 8,
                       &Bs[buf][row][sub * 8]);
        }
    };

    v8f acc[2][4];
#pragma unroll
    for (int i = 0; i < 2; ++i)
#pragma unroll
        for (int j = 0; j < 4; ++j)
#pragma unroll
            for (int r = 0; r < 8; ++r) acc[i][j][r] = 0.0f;

    const int nk = K / 32;
    issue_slab(0, 0);
    for (int it = 0; it < nk; ++it) {
        const int cur = it & 1;
        if (it + 1 < nk) {
            issue_slab(cur ^ 1, (it + 1) * 32);
            // keep the 4 just-issued copies in flight; drain the older 4
            asm volatile("s_wait_asynccnt 0x4" ::: "memory");
        } else {
            asm volatile("s_wait_asynccnt 0x0" ::: "memory");
        }
        __syncthreads();  // slab[cur] fully in LDS for all waves

        v16bf a[2], b[4];
#pragma unroll
        for (int i = 0; i < 2; ++i)
            a[i] = a_frag(&As[cur][wm * 32 + i * 16 + l16][0], half);
#pragma unroll
        for (int j = 0; j < 4; ++j)
            b[j] = b_frag(&Bs[cur][wn * 64 + j * 16 + l16][0], half);
#pragma unroll
        for (int j = 0; j < 4; ++j)
#pragma unroll
            for (int i = 0; i < 2; ++i)
                acc[i][j] = __builtin_amdgcn_wmma_f32_16x16x32_bf16(
                    false, a[i], false, b[j], (short)0, acc[i][j], false, false);

        __syncthreads();  // done reading slab[cur] before it is overwritten
    }

    // D layout: lane L -> col n = L%16, rows m = r + 8*(L/16)
#pragma unroll
    for (int i = 0; i < 2; ++i)
#pragma unroll
        for (int j = 0; j < 4; ++j)
#pragma unroll
            for (int r = 0; r < 8; ++r) {
                const size_t row = (size_t)(m0 + wm * 32 + i * 16 + r + 8 * half);
                Cp[row * N + n0 + wn * 64 + j * 16 + l16] = acc[i][j][r];
            }
}

// ---------- RoPE + cache scatter + bf16 staging ----------
__global__ __launch_bounds__(256) void rope_scatter_kernel(
    const float* __restrict__ qkv, const float* __restrict__ cosp,
    const float* __restrict__ sinp, const int* __restrict__ slots,
    float* __restrict__ key_out, float* __restrict__ val_out,
    __bf16* __restrict__ q_bf, __bf16* __restrict__ k_bf,
    __bf16* __restrict__ vT) {
    const int tid = blockIdx.x * blockDim.x + threadIdx.x;
    const int t = tid >> 7;   // token
    const int d = tid & 127;  // rotary pair index
    if (t >= T_) return;

    const float c = cosp[t * 128 + d];
    const float s = sinp[t * 128 + d];
    const float* row = qkv + (size_t)t * NQKV_;

#pragma unroll
    for (int h = 0; h < H_; ++h) {
        float x1 = row[h * 256 + d];
        float x2 = row[h * 256 + 128 + d];
        q_bf[(size_t)t * 2048 + h * 256 + d]       = (__bf16)(x1 * c - x2 * s);
        q_bf[(size_t)t * 2048 + h * 256 + 128 + d] = (__bf16)(x2 * c + x1 * s);
    }
    const int slot = slots[t];
    {
        float x1 = row[2048 + d];
        float x2 = row[2048 + 128 + d];
        float r1 = x1 * c - x2 * s;
        float r2 = x2 * c + x1 * s;
        key_out[(size_t)slot * 256 + d]       = r1;
        key_out[(size_t)slot * 256 + 128 + d] = r2;
        k_bf[(size_t)t * 256 + d]       = (__bf16)r1;
        k_bf[(size_t)t * 256 + 128 + d] = (__bf16)r2;
    }
    {
        float v1 = row[2304 + d];
        float v2 = row[2304 + 128 + d];
        val_out[(size_t)slot * 256 + d]       = v1;
        val_out[(size_t)slot * 256 + 128 + d] = v2;
        const int b = t >> 11, sx = t & 2047;
        vT[((size_t)(b * 256) + d) * (size_t)S_ + sx]       = (__bf16)v1;
        vT[((size_t)(b * 256) + d + 128) * (size_t)S_ + sx] = (__bf16)v2;
    }
}

// ---------- flash attention: 1 wave per (b, h, 16-row q tile) ----------
__global__ __launch_bounds__(128) void attn_kernel(
    const __bf16* __restrict__ q_bf, const __bf16* __restrict__ k_bf,
    const __bf16* __restrict__ vT, __bf16* __restrict__ attn) {
    __shared__ alignas(16) __bf16 Pb[4][16][32];  // per-wave P tile

    const int lane = threadIdx.x & 31;
    const int wid  = threadIdx.x >> 5;
    const int half = lane >> 4;
    const int l16  = lane & 15;

    const int gw = blockIdx.x * 4 + wid;
    const int qt = gw & 127;
    const int hh = (gw >> 7) & 7;
    const int b  = gw >> 10;
    const int q0 = qt * 16;

    v16bf aQ[8];
    {
        const __bf16* qrow =
            q_bf + (size_t)(b * S_ + q0 + l16) * 2048 + hh * 256;
#pragma unroll
        for (int dd = 0; dd < 8; ++dd) aQ[dd] = a_frag(qrow + dd * 32, half);
    }

    v8f o[16];
#pragma unroll
    for (int t = 0; t < 16; ++t)
#pragma unroll
        for (int r = 0; r < 8; ++r) o[t][r] = 0.0f;

    float mrun[8], lrun[8];
#pragma unroll
    for (int r = 0; r < 8; ++r) { mrun[r] = -3.0e38f; lrun[r] = 0.0f; }

    const float scale = 0.0625f;  // D^-0.5
    const int kend = q0 + 16;

    for (int k0 = 0; k0 < kend; k0 += 32) {
        v8f sc[2];
#pragma unroll
        for (int t2 = 0; t2 < 2; ++t2)
#pragma unroll
            for (int r = 0; r < 8; ++r) sc[t2][r] = 0.0f;

#pragma unroll
        for (int t2 = 0; t2 < 2; ++t2) {
            const int kb = k0 + t2 * 16;
            if (kb < kend) {  // wave-uniform skip of fully-masked tile
                const __bf16* kp =
                    k_bf + (size_t)(b * S_ + kb + l16) * 256 + 16 * half;
#pragma unroll
                for (int dd = 0; dd < 8; ++dd) {
                    v16bf bb = *(const v16bf*)(kp + dd * 32);
                    sc[t2] = __builtin_amdgcn_wmma_f32_16x16x32_bf16(
                        false, aQ[dd], false, bb, (short)0, sc[t2], false,
                        false);
                }
            }
        }

        float corr[8];
#pragma unroll
        for (int r = 0; r < 8; ++r) {
            const int q_row = q0 + r + 8 * half;
            float s0 = sc[0][r] * scale;
            float s1 = sc[1][r] * scale;
            if (k0 + l16 > q_row) s0 = -3.0e38f;
            if (k0 + 16 + l16 > q_row) s1 = -3.0e38f;
            float vmax = fmaxf(s0, s1);
#pragma unroll
            for (int dx = 1; dx < 16; dx <<= 1)
                vmax = fmaxf(vmax, __shfl_xor(vmax, dx, 32));
            const float mnew = fmaxf(mrun[r], vmax);
            const float p0 = __expf(s0 - mnew);
            const float p1 = __expf(s1 - mnew);
            float rs = p0 + p1;
#pragma unroll
            for (int dx = 1; dx < 16; dx <<= 1) rs += __shfl_xor(rs, dx, 32);
            const float cr = __expf(mrun[r] - mnew);
            lrun[r] = lrun[r] * cr + rs;
            mrun[r] = mnew;
            corr[r] = cr;
            Pb[wid][r + 8 * half][l16]      = (__bf16)p0;
            Pb[wid][r + 8 * half][16 + l16] = (__bf16)p1;
        }
#pragma unroll
        for (int t = 0; t < 16; ++t)
#pragma unroll
            for (int r = 0; r < 8; ++r) o[t][r] *= corr[r];

        asm volatile("s_wait_dscnt 0x0" ::: "memory");  // same-wave LDS RAW

        v16bf aP = a_frag(&Pb[wid][l16][0], half);
        int s0i = k0 + 16 * half;
        if (s0i > S_ - 16) s0i = S_ - 16;  // clamp; masked cols have P==0
#pragma unroll
        for (int t = 0; t < 16; ++t) {
            const __bf16* vp =
                vT + ((size_t)(b * 256) + t * 16 + l16) * (size_t)S_ + s0i;
            v16bf bv = *(const v16bf*)vp;
            o[t] = __builtin_amdgcn_wmma_f32_16x16x32_bf16(
                false, aP, false, bv, (short)0, o[t], false, false);
        }
    }

    float inv[8];
#pragma unroll
    for (int r = 0; r < 8; ++r) inv[r] = 1.0f / lrun[r];
#pragma unroll
    for (int t = 0; t < 16; ++t)
#pragma unroll
        for (int r = 0; r < 8; ++r) {
            const size_t row = (size_t)(b * S_ + q0 + r + 8 * half);
            attn[row * 2048 + hh * 256 + t * 16 + l16] =
                (__bf16)(o[t][r] * inv[r]);
        }
}

// ---------- host launcher ----------
extern "C" void kernel_launch(void* const* d_in, const int* in_sizes, int n_in,
                              void* d_out, int out_size, void* d_ws,
                              size_t ws_size, hipStream_t stream) {
    const float* hidden = (const float*)d_in[0];
    const float* cosp   = (const float*)d_in[1];
    const float* sinp   = (const float*)d_in[2];
    const int*   slots  = (const int*)d_in[3];
    const float* key_in = (const float*)d_in[4];
    const float* val_in = (const float*)d_in[5];
    const float* w_qkv  = (const float*)d_in[6];
    const float* w_o    = (const float*)d_in[7];

    float* out     = (float*)d_out;
    float* key_out = out + (size_t)T_ * HID_;
    float* val_out = key_out + (size_t)NSLOTS_ * D_;

    // workspace layout (~170 MB); hid_bf region is reused for attn_bf
    char* ws = (char*)d_ws;
    float*  qkv_ws  = (float*)(ws);                 // 8192*2560 f32 = 80 MB
    __bf16* q_bf    = (__bf16*)(ws + 83886080);     // 8192*2048 bf16 = 32 MB
    __bf16* k_bf    = (__bf16*)(ws + 117440512);    // 8192*256  bf16 =  4 MB
    __bf16* vT      = (__bf16*)(ws + 121634816);    // 4*256*2048 bf16 = 4 MB
    __bf16* hidattn = (__bf16*)(ws + 125829120);    // 8192*2048 bf16 = 32 MB
    __bf16* wq_bf   = (__bf16*)(ws + 159383552);    // 2560*2048 bf16 = 10 MB
    __bf16* wo_bf   = (__bf16*)(ws + 169869312);    // 2048*2048 bf16 =  8 MB

    // 0) pre-convert fp32 operands to bf16 (one pass, memory-bound)
    cvt_bf16_kernel<<<(T_ * HID_ / 4 + 255) / 256, 256, 0, stream>>>(
        hidden, hidattn, T_ * HID_);
    cvt_bf16_kernel<<<(NQKV_ * HID_ / 4 + 255) / 256, 256, 0, stream>>>(
        w_qkv, wq_bf, NQKV_ * HID_);
    cvt_bf16_kernel<<<(HID_ * HID_ / 4 + 255) / 256, 256, 0, stream>>>(
        w_o, wo_bf, HID_ * HID_);

    // 1) initialize full caches from (zero) inputs; scatter overwrites slots
    (void)hipMemcpyAsync(key_out, key_in,
                         (size_t)NSLOTS_ * D_ * sizeof(float),
                         hipMemcpyDeviceToDevice, stream);
    (void)hipMemcpyAsync(val_out, val_in,
                         (size_t)NSLOTS_ * D_ * sizeof(float),
                         hipMemcpyDeviceToDevice, stream);

    // 2) QKV projection: qkv = hidden @ w_qkv^T
    dim3 g1(NQKV_ / 128, T_ / 128);  // (20, 64)
    gemm_nt_bf16_kernel<<<g1, 256, 0, stream>>>(hidattn, wq_bf, qkv_ws,
                                                T_, NQKV_, HID_);

    // 3) RoPE + cache scatter + bf16 staging (q, k, v^T)
    rope_scatter_kernel<<<(T_ * 128) / 256, 256, 0, stream>>>(
        qkv_ws, cosp, sinp, slots, key_out, val_out, q_bf, k_bf, vT);

    // 4) causal flash attention (writes attn into the hid_bf region)
    attn_kernel<<<(B_ * H_ * (S_ / 16)) / 4, 128, 0, stream>>>(q_bf, k_bf, vT,
                                                               hidattn);

    // 5) output projection: out = attn @ w_o^T
    dim3 g2(HID_ / 128, T_ / 128);  // (16, 64)
    gemm_nt_bf16_kernel<<<g2, 256, 0, stream>>>(hidattn, wo_bf, out,
                                                T_, HID_, HID_);
}
